// drug_graph_module_49718541419112
// MI455X (gfx1250) — matrix operand order repaired and checked by reading, not compiled, not run on hardware
//
#include <hip/hip_runtime.h>
#include <hip/hip_bf16.h>

// ---------------------------------------------------------------------------
// Types for CDNA5 WMMA (wave32, v_wmma_f32_16x16x32_bf16)
// ---------------------------------------------------------------------------
typedef __attribute__((ext_vector_type(16))) __bf16 v16bf;
typedef __attribute__((ext_vector_type(2)))  __bf16 v2bf;
typedef __attribute__((ext_vector_type(8)))  float  v8f;
typedef __attribute__((ext_vector_type(8)))  unsigned short v8us;
typedef __attribute__((ext_vector_type(4)))  int    v4i;

#define AS1 __attribute__((address_space(1)))
#define AS3 __attribute__((address_space(3)))

struct u16x16 { v8us lo, hi; };

__device__ __forceinline__ unsigned short f2bf(float f) {
    // round-to-nearest-even fp32 -> bf16 (bit math; no __bf16 arithmetic)
    unsigned u = __float_as_uint(f);
    return (unsigned short)((u + 0x7FFFu + ((u >> 16) & 1u)) >> 16);
}

__device__ __forceinline__ unsigned pack2bf(float a, float b) {
#if __has_builtin(__builtin_amdgcn_cvt_pk_bf16_f32)
    v2bf r = __builtin_amdgcn_cvt_pk_bf16_f32(a, b);   // single VALU op
    return __builtin_bit_cast(unsigned, r);
#else
    return (unsigned)f2bf(a) | ((unsigned)f2bf(b) << 16);
#endif
}

__device__ __forceinline__ v16bf mkfrag(const unsigned short* p0,
                                        const unsigned short* p1) {
    u16x16 t;
    t.lo = *(const v8us*)p0;   // 16-byte LDS load
    t.hi = *(const v8us*)p1;   // 16-byte LDS load
    return __builtin_bit_cast(v16bf, t);
}

__device__ __forceinline__ void atomAddF32(float* p, float v) {
#if __has_builtin(__builtin_amdgcn_global_atomic_fadd_f32)
    __builtin_amdgcn_global_atomic_fadd_f32(p, v);
#else
    atomicAdd(p, v);
#endif
}

__device__ __forceinline__ void waitAsync0() {
#if __has_builtin(__builtin_amdgcn_s_wait_asynccnt)
    __builtin_amdgcn_s_wait_asynccnt(0);
#else
    asm volatile("s_wait_asynccnt 0" ::: "memory");
#endif
}

#define TM 128   // rows per block
#define KC 32    // K-chunk (matches 16x16x32 bf16 WMMA)

// ---------------------------------------------------------------------------
// Shared compute: one 16-row wave strip x 128 cols, 8 WMMAs on staged tiles.
// A: [128 rows][KC k] bf16 bits.  B: [128 cols][bstride k] bf16 bits.
// ---------------------------------------------------------------------------
__device__ __forceinline__ void wmma_tile(const unsigned short* __restrict__ A,
                                          const unsigned short* __restrict__ B,
                                          int bstride, int bkoff,
                                          int lane, int r0, v8f (&acc)[8]) {
    // A fragment (16x32 bf16): lanes 0-15 row M=lane, K 0..7 & 16..23;
    // lanes 16-31 same rows, K 8..15 & 24..31.
    const int mrow = r0 + (lane & 15);
    const int ksel = (lane >> 4) * 8;
    v16bf afrag = mkfrag(&A[mrow * KC + ksel], &A[mrow * KC + 16 + ksel]);
    const int kselB = (lane >> 4) * 8;
#pragma unroll
    for (int c = 0; c < 8; ++c) {
        int n = c * 16 + (lane & 15);
        v16bf bfrag = mkfrag(&B[n * bstride + bkoff + kselB],
                             &B[n * bstride + bkoff + kselB + 16]);
        acc[c] = __builtin_amdgcn_wmma_f32_16x16x32_bf16(
            false, afrag, false, bfrag, (short)0, acc[c], false, false);
    }
}

// ---------------------------------------------------------------------------
// Big-K GEMM staging: A tile 128x32 from X[N,K] (K arbitrary stride, b32 loads).
// Thread owns fixed (colpair c2, row r00), rows advance by 16 -> constant
// address strides, branch-free guards via clamp+select.
// ---------------------------------------------------------------------------
template<bool RG, bool KG>
__device__ __forceinline__ void loadA_g(const float* __restrict__ X,
                                        int rowBase, int k0, int nrows, int K,
                                        int tid, float (&a0)[8], float (&a1)[8]) {
    const int c2  = (tid & 15) * 2;
    const int r00 = tid >> 4;
    const int g0  = k0 + c2, g1 = g0 + 1;
    const int cc0 = KG ? (g0 < K ? g0 : K - 1) : g0;
    const int cc1 = KG ? (g1 < K ? g1 : K - 1) : g1;
#pragma unroll
    for (int j = 0; j < 8; ++j) {
        int gr  = rowBase + r00 + 16 * j;
        int grc = RG ? (gr < nrows ? gr : 0) : gr;
        const float* p = X + (long long)grc * K;
        float v0 = p[cc0];
        float v1 = p[cc1];
        if (KG) { if (g0 >= K) v0 = 0.0f; if (g1 >= K) v1 = 0.0f; }
        if (RG && gr >= nrows) { v0 = 0.0f; v1 = 0.0f; }
        a0[j] = v0; a1[j] = v1;
    }
}

__device__ __forceinline__ void storeA_l(unsigned short* __restrict__ Alds, int tid,
                                         const float (&a0)[8], const float (&a1)[8]) {
    const int c2  = (tid & 15) * 2;
    const int r00 = tid >> 4;
#pragma unroll
    for (int j = 0; j < 8; ++j) {
        int row = r00 + 16 * j;
        *(unsigned*)&Alds[row * KC + c2] = pack2bf(a0[j], a1[j]);
    }
}

// B tile for big-K: W rows k0..k0+31, 128 cols, stored transposed [col][k].
template<bool KG>
__device__ __forceinline__ void loadB_g(const float* __restrict__ W, int k0, int K,
                                        int tid, float (&w0)[8], float (&w1)[8]) {
    const int col = tid & 127;
    const int kb  = (tid >> 7) * 16;
#pragma unroll
    for (int j = 0; j < 8; ++j) {
        int kk = kb + 2 * j;
        int g0 = k0 + kk, g1 = g0 + 1;
        int c0 = KG ? (g0 < K ? g0 : K - 1) : g0;
        int c1 = KG ? (g1 < K ? g1 : K - 1) : g1;
        float v0 = W[(long long)c0 * 128 + col];
        float v1 = W[(long long)c1 * 128 + col];
        if (KG) { if (g0 >= K) v0 = 0.0f; if (g1 >= K) v1 = 0.0f; }
        w0[j] = v0; w1[j] = v1;
    }
}

__device__ __forceinline__ void storeB_l(unsigned short* __restrict__ Blds, int tid,
                                         const float (&w0)[8], const float (&w1)[8]) {
    const int col = tid & 127;
    const int kb  = (tid >> 7) * 16;
#pragma unroll
    for (int j = 0; j < 8; ++j) {
        int kk = kb + 2 * j;
        *(unsigned*)&Blds[col * KC + kk] = pack2bf(w0[j], w1[j]);
    }
}

// ---------------------------------------------------------------------------
// Layer-0 projection: P[N,128] = X[N,K] @ W[K,128], K = 5181.
// Double-buffered LDS, register-staged software pipeline, one barrier/iter.
// ---------------------------------------------------------------------------
template<bool RG>
__global__ __launch_bounds__(256)
void gemm_bigK(const float* __restrict__ X, const float* __restrict__ W,
               float* __restrict__ P, int nrows, int K, int rowBase0) {
    __shared__ __align__(16) unsigned short Alds[2][TM * KC];
    __shared__ __align__(16) unsigned short Blds[2][128 * KC];

    const int tid  = threadIdx.x;
    const int lane = tid & 31;
    const int r0   = (tid >> 5) * 16;
    const int rowBase = rowBase0 + blockIdx.x * TM;

    v8f acc[8];
#pragma unroll
    for (int c = 0; c < 8; ++c) acc[c] = (v8f)(0.0f);

    float a0[8], a1[8], w0[8], w1[8];
    loadA_g<RG, false>(X, rowBase, 0, nrows, K, tid, a0, a1);
    loadB_g<false>(W, 0, K, tid, w0, w1);
    storeA_l(Alds[0], tid, a0, a1);
    storeB_l(Blds[0], tid, w0, w1);

    const int nkt = (K + KC - 1) / KC;
    for (int kt = 0; kt < nkt; ++kt) {
        __syncthreads();
        const int cur  = kt & 1;
        const bool more = (kt + 1 < nkt);
        if (more) {
            const int k0n = (kt + 1) * KC;
            if (k0n + KC <= K) {
                loadA_g<RG, false>(X, rowBase, k0n, nrows, K, tid, a0, a1);
                loadB_g<false>(W, k0n, K, tid, w0, w1);
            } else {
                loadA_g<RG, true>(X, rowBase, k0n, nrows, K, tid, a0, a1);
                loadB_g<true>(W, k0n, K, tid, w0, w1);
            }
        }
        wmma_tile(Alds[cur], Blds[cur], KC, 0, lane, r0, acc);
        if (more) {
            storeA_l(Alds[cur ^ 1], tid, a0, a1);
            storeB_l(Blds[cur ^ 1], tid, w0, w1);
        }
    }

    // C/D layout: VGPR v, lanes 0-15 -> M=v, lanes 16-31 -> M=v+8; N=lane&15
#pragma unroll
    for (int c = 0; c < 8; ++c) {
        int col = c * 16 + (lane & 15);
#pragma unroll
        for (int v = 0; v < 8; ++v) {
            int gr = rowBase + r0 + v + (lane >> 4) * 8;
            if (!RG || gr < nrows) P[(long long)gr * 128 + col] = acc[c][v];
        }
    }
}

// ---------------------------------------------------------------------------
// K=128 GEMM against a pre-converted transposed bf16 weight Wt[col][k].
// FUSED:  A_eff = relu(A + A2 + abias);  Out = relu(A_eff @ W + obias)
// !FUSED: Out = A @ W
// ---------------------------------------------------------------------------
template<bool FUSED, bool RG>
__device__ __forceinline__ void loadA128(const float* __restrict__ A,
                                         const float* __restrict__ A2,
                                         const float* __restrict__ abias,
                                         int rowBase, int k0, int nrows,
                                         int tid, float2 (&av)[8]) {
    const int c2  = (tid & 15) * 2;
    const int r00 = tid >> 4;
    const int gc  = k0 + c2;
    float bb0 = 0.0f, bb1 = 0.0f;
    if (FUSED) { bb0 = abias[gc]; bb1 = abias[gc + 1]; }
#pragma unroll
    for (int j = 0; j < 8; ++j) {
        int gr  = rowBase + r00 + 16 * j;
        int grc = RG ? (gr < nrows ? gr : 0) : gr;
        long long base = (long long)grc * 128 + gc;
        float2 v = *(const float2*)&A[base];         // 8B aligned (gc even)
        if (FUSED) {
            float2 g = *(const float2*)&A2[base];
            v.x = fmaxf(v.x + g.x + bb0, 0.0f);
            v.y = fmaxf(v.y + g.y + bb1, 0.0f);
        }
        if (RG && gr >= nrows) { v.x = 0.0f; v.y = 0.0f; }
        av[j] = v;
    }
}

__device__ __forceinline__ void storeA128(unsigned short* __restrict__ Alds, int tid,
                                          const float2 (&av)[8]) {
    const int c2  = (tid & 15) * 2;
    const int r00 = tid >> 4;
#pragma unroll
    for (int j = 0; j < 8; ++j) {
        int row = r00 + 16 * j;
        *(unsigned*)&Alds[row * KC + c2] = pack2bf(av[j].x, av[j].y);
    }
}

template<bool FUSED, bool RG>
__global__ __launch_bounds__(256)
void gemm128k(const float* __restrict__ A, const float* __restrict__ A2,
              const float* __restrict__ abias,
              const unsigned short* __restrict__ Wt,   // [128 col][128 k] bf16
              const float* __restrict__ obias,
              float* __restrict__ Out, int nrows, int rowBase0) {
    __shared__ __align__(16) unsigned short Alds[2][TM * KC];
    __shared__ __align__(16) unsigned short Bl[128 * 128];   // full weight, 32KB

    const int tid  = threadIdx.x;
    const int lane = tid & 31;
    const int r0   = (tid >> 5) * 16;
    const int rowBase = rowBase0 + blockIdx.x * TM;

    // Stage whole transposed weight once: contiguous 32KB global -> LDS.
#if __has_builtin(__builtin_amdgcn_global_load_async_to_lds_b128)
    {
        // CDNA5 async copy (ASYNCcnt): no VGPR round-trip, no ds_store.
        AS1 v4i* sg = (AS1 v4i*)Wt;
        AS3 v4i* dl = (AS3 v4i*)Bl;
#pragma unroll
        for (int j = 0; j < 8; ++j) {
            int idx = tid + 256 * j;            // 16B-granule index, 2048 total
            __builtin_amdgcn_global_load_async_to_lds_b128(sg + idx, dl + idx, 0, 0);
        }
        waitAsync0();
    }
#else
    {
        const uint4* s = (const uint4*)Wt;
        uint4*       d = (uint4*)Bl;
#pragma unroll
        for (int j = 0; j < 8; ++j) d[tid + 256 * j] = s[tid + 256 * j];
    }
#endif

    v8f acc[8];
#pragma unroll
    for (int c = 0; c < 8; ++c) acc[c] = (v8f)(0.0f);

    float2 av[8];
    loadA128<FUSED, RG>(A, A2, abias, rowBase, 0, nrows, tid, av);
    storeA128(Alds[0], tid, av);

#pragma unroll
    for (int kt = 0; kt < 4; ++kt) {       // K = 128 = 4 * 32
        __syncthreads();
        const int cur = kt & 1;
        if (kt < 3)
            loadA128<FUSED, RG>(A, A2, abias, rowBase, (kt + 1) * KC, nrows, tid, av);
        wmma_tile(Alds[cur], Bl, 128, kt * KC, lane, r0, acc);
        if (kt < 3) storeA128(Alds[cur ^ 1], tid, av);
    }

#pragma unroll
    for (int c = 0; c < 8; ++c) {
        int col = c * 16 + (lane & 15);
        float b = FUSED ? obias[col] : 0.0f;
#pragma unroll
        for (int v = 0; v < 8; ++v) {
            int gr = rowBase + r0 + v + (lane >> 4) * 8;
            if (!RG || gr < nrows) {
                float val = acc[c][v] + b;
                if (FUSED) val = fmaxf(val, 0.0f);
                Out[(long long)gr * 128 + col] = val;
            }
        }
    }
}

// ---------------------------------------------------------------------------
// Weight prep: fp32 W[k][col] (128x128) -> bf16 Wt[col][k] (transposed).
// ---------------------------------------------------------------------------
__global__ void prep_weight(const float* __restrict__ W,
                            unsigned short* __restrict__ Wt) {
    int idx = blockIdx.x * 256 + threadIdx.x;    // 16384 elems, grid 64
    int col = idx & 127, k = idx >> 7;
    Wt[col * 128 + k] = f2bf(W[k * 128 + col]);
}

// ---------------------------------------------------------------------------
// Utility + graph kernels
// ---------------------------------------------------------------------------
__global__ void fill_zero(float* __restrict__ p, int n) {
    int i = blockIdx.x * 256 + threadIdx.x;
    if (i < n) p[i] = 0.0f;
}

// AGG[dst] += P[src] over all edges, 4 channels per thread (float4 gather).
__global__ void edge_scatter(const float* __restrict__ P,
                             const int* __restrict__ src,
                             const int* __restrict__ dst,
                             float* __restrict__ AGG, int nE) {
    long long gid = (long long)blockIdx.x * 256 + threadIdx.x;
    int e = (int)(gid >> 5);
    if (e >= nE) return;
    int c4 = ((int)gid & 31) * 4;
    int s = src[e], d = dst[e];
    const float4 v = *(const float4*)&P[(long long)s * 128 + c4];
    float* pd = &AGG[(long long)d * 128 + c4];
    atomAddF32(pd + 0, v.x);
    atomAddF32(pd + 1, v.y);
    atomAddF32(pd + 2, v.z);
    atomAddF32(pd + 3, v.w);
}

// Per-channel sum / sumsq over all rows (block-local reduce, then atomics).
__global__ __launch_bounds__(256)
void bn_stats(const float* __restrict__ S, float* __restrict__ stats, int nrows) {
    __shared__ float red[512];
    const int tid  = threadIdx.x;
    const int c    = tid & 127;
    const int half = tid >> 7;
    float s = 0.0f, sq = 0.0f;
#pragma unroll 4
    for (int i = 0; i < 128; ++i) {
        int r = blockIdx.x * 256 + half + 2 * i;
        if (r < nrows) {
            float v = S[(long long)r * 128 + c];
            s += v; sq += v * v;
        }
    }
    red[tid] = s;
    red[256 + tid] = sq;
    __syncthreads();
    if (tid < 128) {
        atomAddF32(&stats[c],       red[tid] + red[tid + 128]);
        atomAddF32(&stats[128 + c], red[256 + tid] + red[256 + tid + 128]);
    }
}

// Normalize, write H (input to next layer) and gather every-7th row to out.
__global__ void bn_apply(const float* __restrict__ S,
                         const float* __restrict__ stats,
                         const float* __restrict__ gamma,
                         const float* __restrict__ beta,
                         float* __restrict__ H, float* __restrict__ Out,
                         int layer, int nrows) {
    long long idx = (long long)blockIdx.x * 256 + threadIdx.x;
    if (idx >= (long long)nrows * 128) return;
    int r = (int)(idx >> 7), c = (int)(idx & 127);
    float invN = 1.0f / (float)nrows;
    float mean = stats[c] * invN;
    float var  = stats[128 + c] * invN - mean * mean;
    float inv  = rsqrtf(var + 1e-5f);
    float v = (S[idx] - mean) * inv * gamma[c] + beta[c];
    H[idx] = v;
    if (r % 7 == 0)
        Out[(long long)(r / 7) * 384 + layer * 128 + c] = v;
}

// ---------------------------------------------------------------------------
// Host launch
// ---------------------------------------------------------------------------
extern "C" void kernel_launch(void* const* d_in, const int* in_sizes, int n_in,
                              void* d_out, int out_size, void* d_ws, size_t ws_size,
                              hipStream_t stream) {
    const int N = 50000, E = 200000, K0 = 5181;

    const float* x  = (const float*)d_in[0];
    const int*   ei = (const int*)d_in[1];
    // d_in[2] = batch (unused: single graph, output is a fixed row gather)
    const float* W1[3] = {(const float*)d_in[3], (const float*)d_in[9],  (const float*)d_in[15]};
    const float* b1[3] = {(const float*)d_in[4], (const float*)d_in[10], (const float*)d_in[16]};
    const float* W2[3] = {(const float*)d_in[5], (const float*)d_in[11], (const float*)d_in[17]};
    const float* b2[3] = {(const float*)d_in[6], (const float*)d_in[12], (const float*)d_in[18]};
    const float* gm[3] = {(const float*)d_in[7], (const float*)d_in[13], (const float*)d_in[19]};
    const float* bt[3] = {(const float*)d_in[8], (const float*)d_in[14], (const float*)d_in[20]};

    const int* src = ei;
    const int* dst = ei + E;

    float* ws   = (float*)d_ws;
    size_t nd   = (size_t)N * 128;
    float* P    = ws;             // [N,128] projected features
    float* AGG  = P + nd;         // [N,128] neighbor sums
    float* S    = AGG + nd;       // [N,128] pre-BN activations
    float* H    = S + nd;         // [N,128] normalized (next-layer input)
    float* stats = H + nd;        // [256]   sum / sumsq per channel
    unsigned short* wtp = (unsigned short*)(stats + 256);
    unsigned short* WtW2[3]  = {wtp, wtp + 16384, wtp + 32768};
    unsigned short* WtW1n[2] = {wtp + 49152, wtp + 65536};   // W1 of layers 1,2

    float* out = (float*)d_out;

    dim3 blk(256);
    const int mainBlocks = N / TM;            // 390
    const int tailBase   = mainBlocks * TM;   // 49920 (80-row guarded tail)
    const int fillGrid = (int)((nd + 255) / 256);
    const int edgeGrid = (int)(((long long)E * 32 + 255) / 256);
    const int bnsGrid  = (N + 255) / 256;
    const int bnaGrid  = (int)((nd + 255) / 256);

    // One-time weight prep (bf16, transposed) for the 128x128 GEMMs.
    for (int i = 0; i < 3; ++i)
        prep_weight<<<64, blk, 0, stream>>>(W2[i], WtW2[i]);
    prep_weight<<<64, blk, 0, stream>>>(W1[1], WtW1n[0]);
    prep_weight<<<64, blk, 0, stream>>>(W1[2], WtW1n[1]);

    // Layer 0 projection in 128-dim space (GIN agg is linear => project first)
    gemm_bigK<false><<<mainBlocks, blk, 0, stream>>>(x, W1[0], P, N, K0, 0);
    gemm_bigK<true ><<<1,          blk, 0, stream>>>(x, W1[0], P, N, K0, tailBase);

    for (int i = 0; i < 3; ++i) {
        fill_zero<<<fillGrid, blk, 0, stream>>>(AGG, (int)nd);
        edge_scatter<<<edgeGrid, blk, 0, stream>>>(P, src, dst, AGG, E);
        fill_zero<<<1, blk, 0, stream>>>(stats, 256);
        // S = relu( relu(P + AGG + b1) @ W2 + b2 )
        gemm128k<true, false><<<mainBlocks, blk, 0, stream>>>(
            P, AGG, b1[i], WtW2[i], b2[i], S, N, 0);
        gemm128k<true, true><<<1, blk, 0, stream>>>(
            P, AGG, b1[i], WtW2[i], b2[i], S, N, tailBase);
        bn_stats<<<bnsGrid, blk, 0, stream>>>(S, stats, N);
        bn_apply<<<bnaGrid, blk, 0, stream>>>(S, stats, gm[i], bt[i],
                                              H, out, i, N);
        if (i < 2) {
            // P = H @ W1[i+1]  (next layer projection)
            gemm128k<false, false><<<mainBlocks, blk, 0, stream>>>(
                H, nullptr, nullptr, WtW1n[i], nullptr, P, N, 0);
            gemm128k<false, true><<<1, blk, 0, stream>>>(
                H, nullptr, nullptr, WtW1n[i], nullptr, P, N, tailBase);
        }
    }
}
